// InfoNceLoss_33354716021589
// MI455X (gfx1250) — compile-verified, matrix operand order checked by reading, not run on hardware
//
#include <hip/hip_runtime.h>

#define N_ROWS 8192
#define DIM    1024
#define TEMP_INV 10.0f
#define EPS_F  1e-8f
#define NEG_INF (-3.402823466e38f)

typedef unsigned short u16;
typedef __attribute__((ext_vector_type(16))) __bf16 v16bf;
typedef __attribute__((ext_vector_type(8)))  float  v8f;

// ---------------------------------------------------------------------------
// Fragment loader (16-bit 16x32 A layout, ISA 7.12.2):
//   lane = m + 16*kh; VGPR0..3 <- K = kb + kh*8 + {0..7}  (16 contiguous bytes)
//                     VGPR4..7 <- K = kb + 16 + kh*8 + {0..7}
// For a Gram matrix E*E^T the B fragment (K-major 32x16) uses the identical
// per-lane pattern on row-major E with m = column index.
// ---------------------------------------------------------------------------
__device__ __forceinline__ v16bf load_frag(const u16* __restrict__ M,
                                           int rowbase, int kb, int m, int kh) {
  const uint4* p = (const uint4*)(M + (size_t)(rowbase + m) * DIM + kb + kh * 8);
  v16bf f;
  ((uint4*)&f)[0] = p[0];
  ((uint4*)&f)[1] = p[2];
  return f;
}

// Same pattern but from an LDS-resident 32xDIM tile (rbase = 0 or 16).
__device__ __forceinline__ v16bf lds_frag(const u16* S, int rbase, int kb,
                                          int m, int kh) {
  const uint4* p = (const uint4*)(S + (rbase + m) * DIM + kb + kh * 8);
  v16bf f;
  ((uint4*)&f)[0] = p[0];
  ((uint4*)&f)[1] = p[2];
  return f;
}

__device__ __forceinline__ v8f wmma_acc(v16bf a, v16bf b, v8f c) {
  // (neg_a, A, neg_b, B, c_mod, C, reuse_a, reuse_b)
  return __builtin_amdgcn_wmma_f32_16x16x32_bf16(false, a, false, b,
                                                 (short)0, c, false, false);
}

// ---------------------------------------------------------------------------
// Kernel 1: row L2-normalize + split fp32 -> (bf16 hi, bf16 lo residual)
// ---------------------------------------------------------------------------
__global__ __launch_bounds__(256) void norm_split_kernel(
    const float* __restrict__ E, u16* __restrict__ Ah, u16* __restrict__ Al) {
  __shared__ float red[256];
  const int row = blockIdx.x;
  const int tid = threadIdx.x;
  const float* x = E + (size_t)row * DIM;

  float ss = 0.f;
#pragma unroll
  for (int i = 0; i < DIM / 256; ++i) {
    float v = x[tid + i * 256];
    ss += v * v;
  }
  red[tid] = ss;
  __syncthreads();
  for (int st = 128; st > 0; st >>= 1) {
    if (tid < st) red[tid] += red[tid + st];
    __syncthreads();
  }
  const float inv = 1.0f / fmaxf(sqrtf(red[0]), 1e-12f);

#pragma unroll
  for (int i = 0; i < DIM / 256; ++i) {
    const int idx = tid + i * 256;
    float v = x[idx] * inv;
    unsigned u  = __float_as_uint(v);
    unsigned hb = u & 0xFFFF0000u;            // truncate to bf16
    float    vh = __uint_as_float(hb);
    float    rs = v - vh;                     // residual, |rs| < 2^-7 |v|
    unsigned lb = __float_as_uint(rs);
    Ah[(size_t)row * DIM + idx] = (u16)(hb >> 16);
    Al[(size_t)row * DIM + idx] = (u16)(lb >> 16);
  }
}

// ---------------------------------------------------------------------------
// Kernel 2: fused Gram-GEMM (bf16x3 WMMA) + exp/mask/row-reduce + per-row loss
// Workgroup (8 waves) owns a 32-row band staged in LDS (hi+lo = 128 KB);
// each wave computes 32x32 output tiles (4 C accumulators) over its strided
// set of column tiles. Per k-step: 8 ds_load_b128 (A) + 8 global_load_b128 (B)
// feed 12 WMMAs.
// ---------------------------------------------------------------------------
__global__ __launch_bounds__(256) void supcon_kernel(
    const u16* __restrict__ Ah, const u16* __restrict__ Al,
    const long long* __restrict__ labels, float* __restrict__ losses) {
  __shared__ __align__(16) u16 sAh[32 * DIM];   // 64 KB
  __shared__ __align__(16) u16 sAl[32 * DIM];   // 64 KB
  __shared__ float sPos[8][32], sAll[8][32], sMax[8][32];

  const int tid  = threadIdx.x;
  const int wave = tid >> 5;
  const int lane = tid & 31;
  const int m    = lane & 15;
  const int kh   = lane >> 4;
  const int rowbase = blockIdx.x * 32;

  // Cooperative stage of the 32-row band (hi+lo) into LDS.
  {
    const uint4* gH = (const uint4*)(Ah + (size_t)rowbase * DIM);
    const uint4* gL = (const uint4*)(Al + (size_t)rowbase * DIM);
    uint4* lH = (uint4*)sAh;
    uint4* lL = (uint4*)sAl;
    for (int i = tid; i < 32 * DIM / 8; i += 256) {
      lH[i] = gH[i];
      lL[i] = gL[i];
    }
  }
  __syncthreads();

  // Row labels for the 16 accumulator slots:
  //   slot r (r<8) -> row r + 8*kh ; slot 8+r -> row 16 + r + 8*kh
  int labr[16];
#pragma unroll
  for (int r = 0; r < 16; ++r) {
    int row = (r & 7) + 8 * kh + ((r >> 3) << 4);
    labr[r] = (int)labels[rowbase + row];
  }

  float pos[16], alls[16], mx[16];
#pragma unroll
  for (int i = 0; i < 16; ++i) { pos[i] = 0.f; alls[i] = 0.f; mx[i] = NEG_INF; }

#define EPILOG(val, grow, gcol, labc, slot)                                  \
  {                                                                          \
    float s_ = (val) * TEMP_INV;                                             \
    if ((grow) != (gcol)) {                                                  \
      float e_ = __expf(s_);                                                 \
      alls[slot] += e_;                                                      \
      mx[slot] = fmaxf(mx[slot], s_);                                        \
      if (labr[slot] == (labc)) pos[slot] += e_;                             \
    }                                                                        \
  }

  for (int jt = wave; jt < N_ROWS / 32; jt += 8) {
    const int colbase = jt * 32;
    const int labc0 = (int)labels[colbase + m];
    const int labc1 = (int)labels[colbase + 16 + m];
    const int gcol0 = colbase + m;
    const int gcol1 = colbase + 16 + m;

    v8f c00 = {0.f,0.f,0.f,0.f,0.f,0.f,0.f,0.f};
    v8f c01 = {0.f,0.f,0.f,0.f,0.f,0.f,0.f,0.f};
    v8f c10 = {0.f,0.f,0.f,0.f,0.f,0.f,0.f,0.f};
    v8f c11 = {0.f,0.f,0.f,0.f,0.f,0.f,0.f,0.f};

    for (int kb = 0; kb < DIM; kb += 32) {
      v16bf b0H = load_frag(Ah, colbase,      kb, m, kh);
      v16bf b0L = load_frag(Al, colbase,      kb, m, kh);
      v16bf b1H = load_frag(Ah, colbase + 16, kb, m, kh);
      v16bf b1L = load_frag(Al, colbase + 16, kb, m, kh);
      v16bf a0H = lds_frag(sAh, 0,  kb, m, kh);
      v16bf a0L = lds_frag(sAl, 0,  kb, m, kh);
      v16bf a1H = lds_frag(sAh, 16, kb, m, kh);
      v16bf a1L = lds_frag(sAl, 16, kb, m, kh);
      // sim = aH*bH + aH*bL + aL*bH  (bf16x3 ~ fp32)
      c00 = wmma_acc(a0H, b0H, c00);
      c00 = wmma_acc(a0H, b0L, c00);
      c00 = wmma_acc(a0L, b0H, c00);
      c01 = wmma_acc(a0H, b1H, c01);
      c01 = wmma_acc(a0H, b1L, c01);
      c01 = wmma_acc(a0L, b1H, c01);
      c10 = wmma_acc(a1H, b0H, c10);
      c10 = wmma_acc(a1H, b0L, c10);
      c10 = wmma_acc(a1L, b0H, c10);
      c11 = wmma_acc(a1H, b1H, c11);
      c11 = wmma_acc(a1H, b1L, c11);
      c11 = wmma_acc(a1L, b1H, c11);
    }

    // C/D layout: lane holds (M = r + 8*kh, N = m) in VGPR r of each C tile.
#pragma unroll
    for (int r = 0; r < 8; ++r) {
      EPILOG(c00[r], rowbase + r + 8 * kh,      gcol0, labc0, r);
      EPILOG(c01[r], rowbase + r + 8 * kh,      gcol1, labc1, r);
      EPILOG(c10[r], rowbase + 16 + r + 8 * kh, gcol0, labc0, 8 + r);
      EPILOG(c11[r], rowbase + 16 + r + 8 * kh, gcol1, labc1, 8 + r);
    }
  }
#undef EPILOG

  // Reduce over the 16 lanes of each half (xor masks < 16 stay in-half).
#pragma unroll
  for (int d = 1; d < 16; d <<= 1) {
#pragma unroll
    for (int i = 0; i < 16; ++i) {
      pos[i]  += __shfl_xor(pos[i],  d, 32);
      alls[i] += __shfl_xor(alls[i], d, 32);
      mx[i]    = fmaxf(mx[i], __shfl_xor(mx[i], d, 32));
    }
  }

  // Lane 0 of each half publishes its 16 row totals (constant indices only).
  if (m == 0) {
#pragma unroll
    for (int r = 0; r < 8; ++r) {
      int row0 = r + 8 * kh;
      int row1 = 16 + r + 8 * kh;
      sPos[wave][row0] = pos[r];      sPos[wave][row1] = pos[8 + r];
      sAll[wave][row0] = alls[r];     sAll[wave][row1] = alls[8 + r];
      sMax[wave][row0] = mx[r];       sMax[wave][row1] = mx[8 + r];
    }
  }
  __syncthreads();

  if (tid < 32) {
    float P = 0.f, A = 0.f, M = NEG_INF;
#pragma unroll
    for (int w = 0; w < 8; ++w) {
      P += sPos[w][tid];
      A += sAll[w][tid];
      M = fmaxf(M, sMax[w][tid]);
    }
    // exp(s) >= exp(-10) > 0, so "has positives" <=> P > 0.
    float loss = (P > 0.f) ? (__logf(A + EPS_F) - __logf(P + EPS_F)) : -M;
    losses[rowbase + tid] = loss;
  }
}

// ---------------------------------------------------------------------------
// Kernel 3: mean over 8192 per-row losses -> scalar
// ---------------------------------------------------------------------------
__global__ __launch_bounds__(256) void mean_kernel(const float* __restrict__ losses,
                                                   float* __restrict__ out) {
  __shared__ float s[256];
  float a = 0.f;
  for (int i = threadIdx.x; i < N_ROWS; i += 256) a += losses[i];
  s[threadIdx.x] = a;
  __syncthreads();
  for (int st = 128; st > 0; st >>= 1) {
    if (threadIdx.x < st) s[threadIdx.x] += s[threadIdx.x + st];
    __syncthreads();
  }
  if (threadIdx.x == 0) out[0] = s[0] * (1.0f / (float)N_ROWS);
}

// ---------------------------------------------------------------------------
extern "C" void kernel_launch(void* const* d_in, const int* in_sizes, int n_in,
                              void* d_out, int out_size, void* d_ws, size_t ws_size,
                              hipStream_t stream) {
  (void)in_sizes; (void)n_in; (void)out_size; (void)ws_size;

  const float*     emb    = (const float*)d_in[0];
  const long long* labels = (const long long*)d_in[1];
  float*           out    = (float*)d_out;

  // Workspace layout: Ah (16 MB bf16) | Al (16 MB bf16) | losses (32 KB f32)
  u16*   Ah     = (u16*)d_ws;
  u16*   Al     = Ah + (size_t)N_ROWS * DIM;
  float* losses = (float*)(Al + (size_t)N_ROWS * DIM);

  norm_split_kernel<<<N_ROWS, 256, 0, stream>>>(emb, Ah, Al);
  supcon_kernel<<<N_ROWS / 32, 256, 0, stream>>>(Ah, Al, labels, losses);
  mean_kernel<<<1, 256, 0, stream>>>(losses, out);
}
